// SimpleHeadAttentionMultiLatent_12601434046470
// MI455X (gfx1250) — matrix operand order repaired and checked by reading, not compiled, not run on hardware
//
#include <hip/hip_runtime.h>

// ---------------------------------------------------------------------------
// Problem constants (from reference)
// ---------------------------------------------------------------------------
constexpr int Bn  = 8;
constexpr int S   = 1024;
constexpr int Dm  = 1024;
constexpr int Hh  = 16;
constexpr int HD  = 64;
constexpr int Lc  = 512;
constexpr int LHD = Lc / Hh;       // 32
constexpr int BS  = Bn * S;        // 8192
constexpr float M_ALPHA = -1.0e9f;

typedef __attribute__((ext_vector_type(16))) _Float16 v16h;
typedef __attribute__((ext_vector_type(8)))  float    v8f;
typedef __attribute__((ext_vector_type(4)))  unsigned u32x4;
typedef __attribute__((ext_vector_type(8)))  int      i32x8;
typedef __attribute__((ext_vector_type(4)))  int      i32x4;

struct __align__(8)  H4 { _Float16 a, b, c, d; };
template<typename T> struct __align__(4 * sizeof(T)) V4 { T a, b, c, d; };

__device__ __forceinline__ _Float16 to_h(float v)    { return (_Float16)v; }
__device__ __forceinline__ _Float16 to_h(_Float16 v) { return v; }

// ---------------------------------------------------------------------------
// Tensor Data Mover (gfx1250): DMA a 2-D tile (row-major, padded) into LDS.
// D# layout per CDNA5 ISA 8.3/8.4. Groups 2/3 zero => tile_dim2/3/4 unused.
// ---------------------------------------------------------------------------
#if defined(__has_builtin)
# if __has_builtin(__builtin_amdgcn_tensor_load_to_lds) && \
     __has_builtin(__builtin_amdgcn_s_wait_tensorcnt)
#  define USE_TDM 1
# endif
#endif

#ifdef USE_TDM
__device__ __forceinline__ void tdm_load_2d(
    unsigned lds_addr, const void* gptr,
    unsigned tile_w /*elems*/, unsigned tile_h /*rows*/,
    unsigned long long row_stride /*elems*/,
    unsigned pad_interval_code, unsigned pad_amount_code) {
  const unsigned long long ga = (unsigned long long)(size_t)gptr;
  u32x4 g0;
  g0[0] = 1u;                                   // count=1, user descriptor
  g0[1] = lds_addr;                             // LDS byte address
  g0[2] = (unsigned)ga;                         // global_addr[31:0]
  g0[3] = (unsigned)((ga >> 32) & 0x01FFFFFFu)  // global_addr[56:32]
          | (2u << 30);                         // type = 2 ("image")
  // group1: data_size=1 (2B), pad_enable, pad geometry, dims/strides
  const unsigned td0 = 1u << 20, td1 = 1u << 20;  // oversized -> no OOB clip
  i32x8 g1;
  g1[0] = (int)((1u << 16) | (1u << 20) |
                (pad_interval_code << 22) | (pad_amount_code << 25));
  g1[1] = (int)((td0 & 0xFFFFu) << 16);         // barrier_addr=0 | dim0.lo
  g1[2] = (int)((td0 >> 16) | ((td1 & 0xFFFFu) << 16));
  g1[3] = (int)((td1 >> 16) | (tile_w << 16));  // dim1.hi | tile_dim0
  g1[4] = (int)(tile_h);                        // tile_dim1 | tile_dim2=0
  g1[5] = (int)(unsigned)row_stride;            // dim0_stride[31:0]
  g1[6] = (int)((unsigned)((row_stride >> 32) & 0xFFFFu)); // stride.hi|d1s.lo
  g1[7] = 0;
  const i32x4 gz = {0, 0, 0, 0};
#if __clang_major__ >= 23
  const i32x8 gz8 = {0, 0, 0, 0, 0, 0, 0, 0};
  __builtin_amdgcn_tensor_load_to_lds(g0, g1, gz, gz, gz8, 0);
#else
  __builtin_amdgcn_tensor_load_to_lds(g0, g1, gz, gz, 0);
#endif
}
#endif  // USE_TDM

// ---------------------------------------------------------------------------
// CDNA5 wave32 WMMA fragment loaders (16x16x32 f16; A/B use contiguous k order, same for both)
//   A 16x32 (MxK): lane L: row = L&15; halves e -> K = (L>>4)*16 + e
//   B 32x16 (KxN): lane L: col = L&15; halves e -> K = (L>>4)*16 + e
//   C/D 16x16 f32: vgpr r, lane L: row = r + 8*(L>>4), col = L&15
// ---------------------------------------------------------------------------
__device__ __forceinline__ v16h ld_afrag(const _Float16* p, int ld) {
  const int lane = threadIdx.x & 31;
  const _Float16* r = p + (lane & 15) * ld + ((lane >> 4) * 16);
  v16h a;
#pragma unroll
  for (int e = 0; e < 8; ++e) { a[e] = r[e]; a[e + 8] = r[e + 8]; }
  return a;
}

// B[k][n] stored row-major p[k*ld + n]
__device__ __forceinline__ v16h ld_bfrag(const _Float16* p, int ld) {
  const int lane = threadIdx.x & 31;
  const int n  = lane & 15;
  const int ko = (lane >> 4) * 16;
  v16h b;
#pragma unroll
  for (int e = 0; e < 16; ++e) b[e] = p[(ko + e) * ld + n];
  return b;
}

// B[k][n] stored transposed p[n*ld + k]  (contiguous along k -> fast LDS read)
__device__ __forceinline__ v16h ld_bfragT(const _Float16* p, int ld) {
  const int lane = threadIdx.x & 31;
  const int n  = lane & 15;
  const int ko = (lane >> 4) * 16;
  v16h b;
#pragma unroll
  for (int e = 0; e < 16; ++e) b[e] = p[n * ld + ko + e];
  return b;
}

__device__ __forceinline__ v8f wmma16(v16h a, v16h b, v8f c) {
  return __builtin_amdgcn_wmma_f32_16x16x32_f16(
      /*neg_a=*/false, a, /*neg_b=*/false, b,
      /*c_mod=*/(short)0, c, /*reuse_a=*/false, /*reuse_b=*/false);
}

// ---------------------------------------------------------------------------
// Tiled WMMA GEMM:  C[M,N] = A[M,K] * B(+bias)
//   TA  : f32 or f16 A source (converted to f16 while staging to LDS)
//   BT  : B source stored as (N,K) row-major (i.e. multiply by B^T)
//   OUT16: write f16 output (else f32 + optional bias)
// Tile: 64x64, BK=32, 128 threads (4 waves); wave w does rows [16w,16w+16).
// ---------------------------------------------------------------------------
template<typename TA, bool BT, bool OUT16>
__global__ __launch_bounds__(128) void gemm_wmma_f16(
    const TA* __restrict__ A, const float* __restrict__ Bsrc,
    const float* __restrict__ bias, float* __restrict__ Cf,
    _Float16* __restrict__ Ch, int M, int N, int K) {
  __shared__ alignas(16) _Float16 As[64][40];
  __shared__ alignas(16) _Float16 Bs[32][72];

  const int tid  = threadIdx.x;
  const int lane = tid & 31;
  const int wave = tid >> 5;
  const int n0 = blockIdx.x * 64;
  const int m0 = blockIdx.y * 64;

  v8f acc[4] = {};

  for (int k0 = 0; k0 < K; k0 += 32) {
    // ---- stage A tile 64x32 (f32/f16 -> f16), 16 elems / thread ----
#pragma unroll
    for (int i = 0; i < 4; ++i) {
      int idx = tid * 16 + i * 4;
      int r = idx >> 5, c = idx & 31;
      V4<TA> v = *(const V4<TA>*)(A + (size_t)(m0 + r) * K + k0 + c);
      As[r][c + 0] = to_h(v.a); As[r][c + 1] = to_h(v.b);
      As[r][c + 2] = to_h(v.c); As[r][c + 3] = to_h(v.d);
    }
    // ---- stage B tile 32x64 (f32 -> f16) ----
#pragma unroll
    for (int i = 0; i < 4; ++i) {
      int idx = tid * 16 + i * 4;
      if (BT) {
        int k = idx & 31, n = idx >> 5;   // contiguous along K in Bsrc(N,K)
        V4<float> v = *(const V4<float>*)(Bsrc + (size_t)(n0 + n) * K + k0 + k);
        Bs[k + 0][n] = to_h(v.a); Bs[k + 1][n] = to_h(v.b);
        Bs[k + 2][n] = to_h(v.c); Bs[k + 3][n] = to_h(v.d);
      } else {
        int k = idx >> 6, n = idx & 63;   // contiguous along N in Bsrc(K,N)
        V4<float> v = *(const V4<float>*)(Bsrc + (size_t)(k0 + k) * N + n0 + n);
        Bs[k][n + 0] = to_h(v.a); Bs[k][n + 1] = to_h(v.b);
        Bs[k][n + 2] = to_h(v.c); Bs[k][n + 3] = to_h(v.d);
      }
    }
    // prefetch next K-slab of A into cache (gfx1250 global_prefetch_b8)
    if (k0 + 32 < K)
      __builtin_prefetch(A + (size_t)(m0 + (tid >> 1)) * K + k0 + 32, 0, 1);

    __syncthreads();

    v16h a = ld_afrag(&As[wave * 16][0], 40);
#pragma unroll
    for (int nt = 0; nt < 4; ++nt) {
      v16h b = ld_bfrag(&Bs[0][nt * 16], 72);
      acc[nt] = wmma16(a, b, acc[nt]);
    }
    __syncthreads();
  }

  // ---- epilogue ----
  const int g = lane >> 4, col = lane & 15;
#pragma unroll
  for (int nt = 0; nt < 4; ++nt) {
#pragma unroll
    for (int r = 0; r < 8; ++r) {
      int row = m0 + wave * 16 + r + 8 * g;
      int cg  = n0 + nt * 16 + col;
      float v = acc[nt][r];
      if (bias) v += bias[cg];
      if constexpr (OUT16) Ch[(size_t)row * N + cg] = (_Float16)v;
      else                 Cf[(size_t)row * N + cg] = v;
    }
  }
}

// ---------------------------------------------------------------------------
// Flash attention, one wave (32 thr) per (b, h, 16-query tile).
//   Q  : (BS, Lc) f16   q_latent, head slice h*LHD..+32
//   Kc : (BS, Lc) f16   combined (= latent K), same slice
//   V  : (BS, Dm) f16   head slice h*HD..+64
//   Z  : (BS, Dm) f16   output, head slice h*HD..+64
// 32 keys per iteration: K/V tiles staged by the Tensor Data Mover
// (tensor_load_to_lds + s_wait_tensorcnt) when available, 2 QK^T WMMAs
// (K=LHD=32), online softmax, P->LDS re-layout, 4 PV WMMAs.
// ---------------------------------------------------------------------------
__global__ __launch_bounds__(32) void mla_flash_attn(
    const _Float16* __restrict__ Q, const _Float16* __restrict__ Kc,
    const _Float16* __restrict__ V, const float* __restrict__ mask,
    _Float16* __restrict__ Z) {
  __shared__ alignas(16) _Float16 Ks[32][40];   // [key][latent], pad 8h = 4 dw
  __shared__ alignas(16) _Float16 Vs[32][72];   // [key][dim],    pad 8h = 4 dw
  __shared__ alignas(16) _Float16 Ps[16][40];   // [row][key]

  const int lane = threadIdx.x;
  const int qt = blockIdx.x, h = blockIdx.y, b = blockIdx.z;
  const int qs = qt * 16;
  const int g = lane >> 4, col = lane & 15;
  const float* mb = mask + (size_t)b * S;

  // Q fragment with 1/sqrt(LHD) folded in
  v16h qf;
  {
    const _Float16* p = Q + ((size_t)b * S + qs + (lane & 15)) * Lc
                          + h * LHD + g * 16;
    const _Float16 sc = (_Float16)0.17677669529663687f;  // 1/sqrt(32)
#pragma unroll
    for (int e = 0; e < 8; ++e) { qf[e] = p[e] * sc; qf[e + 8] = p[e + 8] * sc; }
  }

  float mq[8];
#pragma unroll
  for (int r = 0; r < 8; ++r) mq[r] = mb[qs + r + 8 * g];

  float mrow[8], lrow[8];
#pragma unroll
  for (int r = 0; r < 8; ++r) { mrow[r] = -3.0e38f; lrow[r] = 0.0f; }
  v8f acc[4] = {};

  for (int ks = 0; ks < S; ks += 32) {
    // ---- stage 32 keys of K-latent (32x32) and V (32x64) into LDS ----
#ifdef USE_TDM
    // TDM: pad_interval 16 dw (code 3) / 32 dw (code 4); pad 4 dw (code 3)
    tdm_load_2d((unsigned)(size_t)&Ks[0][0],
                Kc + ((size_t)b * S + ks) * Lc + h * LHD,
                /*w*/LHD, /*h*/32, /*stride*/Lc, /*pi*/3, /*pa*/3);
    tdm_load_2d((unsigned)(size_t)&Vs[0][0],
                V + ((size_t)b * S + ks) * Dm + h * HD,
                /*w*/HD, /*h*/32, /*stride*/Dm, /*pi*/4, /*pa*/3);
    __builtin_amdgcn_s_wait_tensorcnt(0);
#else
    {
      const _Float16* kp = Kc + ((size_t)b * S + ks + lane) * Lc + h * LHD;
#pragma unroll
      for (int c = 0; c < LHD; c += 4)
        *(H4*)&Ks[lane][c] = *(const H4*)(kp + c);
      const _Float16* vp = V + ((size_t)b * S + ks + lane) * Dm + h * HD;
#pragma unroll
      for (int c = 0; c < HD; c += 4)
        *(H4*)&Vs[lane][c] = *(const H4*)(vp + c);
    }
#endif
    __syncthreads();

    // ---- logits: S[16q x 32k] via two WMMAs, B read key-major (transposed) ----
    v16h kf0 = ld_bfragT(&Ks[0][0],  40);
    v16h kf1 = ld_bfragT(&Ks[16][0], 40);
    v8f zc = {};
    v8f s0 = wmma16(qf, kf0, zc);
    v8f s1 = wmma16(qf, kf1, zc);

    const float mk0 = mb[ks + col];
    const float mk1 = mb[ks + 16 + col];

#pragma unroll
    for (int r = 0; r < 8; ++r) {
      float x0 = s0[r] + fminf(1.0f, mq[r] + mk0) * M_ALPHA;
      float x1 = s1[r] + fminf(1.0f, mq[r] + mk1) * M_ALPHA;
      // row max across the 16 lanes that own this row
      float mx = fmaxf(x0, x1);
      mx = fmaxf(mx, __shfl_xor(mx, 1));
      mx = fmaxf(mx, __shfl_xor(mx, 2));
      mx = fmaxf(mx, __shfl_xor(mx, 4));
      mx = fmaxf(mx, __shfl_xor(mx, 8));
      float mnew = fmaxf(mrow[r], mx);
      float f  = __expf(mrow[r] - mnew);
      float p0 = __expf(x0 - mnew);
      float p1 = __expf(x1 - mnew);
      float ps = p0 + p1;
      ps += __shfl_xor(ps, 1);
      ps += __shfl_xor(ps, 2);
      ps += __shfl_xor(ps, 4);
      ps += __shfl_xor(ps, 8);
      lrow[r] = lrow[r] * f + ps;
      mrow[r] = mnew;
#pragma unroll
      for (int nt = 0; nt < 4; ++nt) acc[nt][r] *= f;
      Ps[r + 8 * g][col]      = (_Float16)p0;
      Ps[r + 8 * g][16 + col] = (_Float16)p1;
    }
    __syncthreads();

    // ---- PV: A = P (16x32 keys), B = V tile (32 keys x 4x16 dims) ----
    v16h pa = ld_afrag(&Ps[0][0], 40);
#pragma unroll
    for (int nt = 0; nt < 4; ++nt) {
      v16h vb = ld_bfrag(&Vs[0][nt * 16], 72);
      acc[nt] = wmma16(pa, vb, acc[nt]);
    }
    __syncthreads();
  }

  // ---- normalize and store head slice of Z ----
#pragma unroll
  for (int r = 0; r < 8; ++r) {
    float inv = 1.0f / lrow[r];
    size_t ro = ((size_t)b * S + qs + r + 8 * g) * Dm + h * HD + col;
#pragma unroll
    for (int nt = 0; nt < 4; ++nt)
      Z[ro + nt * 16] = (_Float16)(acc[nt][r] * inv);
  }
}

// ---------------------------------------------------------------------------
// Host-side orchestration
// ---------------------------------------------------------------------------
extern "C" void kernel_launch(void* const* d_in, const int* in_sizes, int n_in,
                              void* d_out, int out_size, void* d_ws, size_t ws_size,
                              hipStream_t stream) {
  const float* x     = (const float*)d_in[0];
  const float* mask  = (const float*)d_in[1];
  const float* wq    = (const float*)d_in[2];
  const float* bq    = (const float*)d_in[3];
  const float* w_ckv = (const float*)d_in[4];
  const float* wuk   = (const float*)d_in[5];   // (Lc, Dm): used transposed
  const float* wuv   = (const float*)d_in[6];   // (Lc, Dm)
  const float* wo    = (const float*)d_in[7];
  const float* bo    = (const float*)d_in[8];
  float* out = (float*)d_out;

  // f16 intermediates in workspace (64 MiB total)
  char* ws = (char*)d_ws;
  _Float16* qfull = (_Float16*)ws;                 ws += (size_t)BS * Dm * 2;
  _Float16* qlat  = (_Float16*)ws;                 ws += (size_t)BS * Lc * 2;
  _Float16* comb  = (_Float16*)ws;                 ws += (size_t)BS * Lc * 2;
  _Float16* vbuf  = (_Float16*)ws;                 ws += (size_t)BS * Dm * 2;
  _Float16* zbuf  = (_Float16*)ws;

  const dim3 blk(128);
  // q_full = x @ wq + bq                       (8192x1024x1024) -> f16
  gemm_wmma_f16<float, false, true><<<dim3(Dm / 64, BS / 64), blk, 0, stream>>>(
      x, wq, bq, nullptr, qfull, BS, Dm, Dm);
  // q_latent = q_full @ wuk^T                  (8192x512x1024)  -> f16
  gemm_wmma_f16<_Float16, true, true><<<dim3(Lc / 64, BS / 64), blk, 0, stream>>>(
      qfull, wuk, nullptr, nullptr, qlat, BS, Lc, Dm);
  // combined = x @ w_ckv                       (8192x512x1024)  -> f16
  gemm_wmma_f16<float, false, true><<<dim3(Lc / 64, BS / 64), blk, 0, stream>>>(
      x, w_ckv, nullptr, nullptr, comb, BS, Lc, Dm);
  // v = combined @ wuv                         (8192x1024x512)  -> f16
  gemm_wmma_f16<_Float16, false, true><<<dim3(Dm / 64, BS / 64), blk, 0, stream>>>(
      comb, wuv, nullptr, nullptr, vbuf, BS, Dm, Lc);
  // flash attention -> zbuf (f16)
  mla_flash_attn<<<dim3(S / 16, Hh, Bn), dim3(32), 0, stream>>>(
      qlat, comb, vbuf, mask, zbuf);
  // out = z @ wo + bo                          (8192x1024x1024) -> f32
  gemm_wmma_f16<_Float16, false, false><<<dim3(Dm / 64, BS / 64), blk, 0, stream>>>(
      zbuf, wo, bo, out, nullptr, BS, Dm, Dm);
}
